// RGT_3118146257187
// MI455X (gfx1250) — compile-verified, matrix operand order checked by reading, not run on hardware
//
#include <hip/hip_runtime.h>
#include <cstdint>
#include <cstddef>

// ---------------------------------------------------------------------------
// Types for gfx1250 WMMA (wave32)
// ---------------------------------------------------------------------------
typedef __attribute__((ext_vector_type(16))) _Float16 v16h;
typedef __attribute__((ext_vector_type(8)))  _Float16 v8h;
typedef __attribute__((ext_vector_type(2)))  _Float16 v2h;
typedef __attribute__((ext_vector_type(8)))  float    v8f;
typedef __attribute__((ext_vector_type(4)))  float    v4f;

#define WAVES_PER_BLOCK 8

// ---------------------------------------------------------------------------
// Float atomic-max via sign-aware integer atomics (init buffer to -1e30f)
// ---------------------------------------------------------------------------
__device__ __forceinline__ void atomicMaxF(float* addr, float val) {
  if (val >= 0.0f) {
    atomicMax((int*)addr, __float_as_int(val));
  } else {
    atomicMin((unsigned int*)addr, __float_as_uint(val));
  }
}

// ---------------------------------------------------------------------------
// Packed 2xf16 global atomic add (CDNA5 GLOBAL_ATOMIC_PK_ADD_F16, no-return).
// ---------------------------------------------------------------------------
__device__ __forceinline__ void atomic_pk_add_f16(_Float16* p, float a, float b) {
  v2h hv; hv[0] = (_Float16)a; hv[1] = (_Float16)b;
  uint32_t bits;
  __builtin_memcpy(&bits, &hv, 4);
  asm volatile("global_atomic_pk_add_f16 %0, %1, off"
               :: "v"(p), "v"(bits) : "memory");
}

// ---------------------------------------------------------------------------
// 8-element vector loads, converted to f16
// ---------------------------------------------------------------------------
__device__ __forceinline__ v8h load8(const _Float16* __restrict__ p) {
  return *(const v8h*)p;                    // one b128 load
}
__device__ __forceinline__ v8h load8(const float* __restrict__ p) {
  v4f a = *(const v4f*)p;                   // two b128 loads
  v4f b = *(const v4f*)(p + 4);
  v8h r;
#pragma unroll
  for (int i = 0; i < 4; ++i) { r[i] = (_Float16)a[i]; r[4 + i] = (_Float16)b[i]; }
  return r;
}

// ---------------------------------------------------------------------------
// A fragment (16x32 f16): lane m=lane&15 holds row m; lanes 0-15 hold
// K 0..7 & 16..23 of the chunk, lanes 16-31 hold K 8..15 & 24..31
// (CDNA5 ISA 7.12.2). K1 is the compile-time concat split: chunk kb reads A1
// if kb < K1 else A2 (chunks never straddle since K1 % 32 == 0).
// ---------------------------------------------------------------------------
template<int K1, int K, typename TIN>
__device__ __forceinline__ v16h load_a_frag(const TIN* __restrict__ A1,
                                            const TIN* __restrict__ A2,
                                            int row0, int kb) {
  const int lane = threadIdx.x & 31;
  const int m    = row0 + (lane & 15);
  const int kh   = lane >> 4;
  const TIN* src;
  if (kb < K1) src = A1 + (size_t)m * K1 + kb;
  else         src = A2 + (size_t)m * (K - K1) + (kb - K1);
  v8h lo = load8(src + kh * 8);
  v8h hi = load8(src + 16 + kh * 8);
  v16h f;
#pragma unroll
  for (int i = 0; i < 8; ++i) { f[i] = lo[i]; f[8 + i] = hi[i]; }
  return f;
}

// ---------------------------------------------------------------------------
// Generic fused GEMM: C[M,NOUT] = act(concat(A1,A2)[M,K] @ W[K,NOUT] + bias)
// Weights staged into LDS pre-swizzled into WMMA B-fragment order: tile
// (kc,ct) occupies 512 halves; lane l's 16 fragment halves are contiguous at
// tile + l*16, so a B fragment is one 32-byte LDS read. The ct loop is kept
// ROLLED so B fragments are re-read from LDS each iteration (cheap) instead
// of being hoisted into registers and spilled to scratch.
// ACT: 0 = none, 1 = ReLU.
// ---------------------------------------------------------------------------
template<typename TIN, typename TOUT, int ACT, int K1, int K, int NOUT>
__global__ void __launch_bounds__(256, 4)
gemm_kernel(const TIN* __restrict__ A1, const TIN* __restrict__ A2,
            const float* __restrict__ W, const float* __restrict__ bias,
            TOUT* __restrict__ C, int M)
{
  constexpr int NK = K / 32;
  constexpr int NT = NOUT / 16;
  __shared__ __align__(64) _Float16 wlds[K * NOUT];
  __shared__ float blds[NOUT];

  const int wave = threadIdx.x >> 5;
  const int lane = threadIdx.x & 31;
  const int mtiles = M >> 4;                 // M multiple of 16 (N=100000)

  // Prefetch this wave's first A tile rows while weights stage through LDS.
  {
    int mt0 = blockIdx.x * WAVES_PER_BLOCK + wave;
    if (mt0 < mtiles) {
      int m = (mt0 << 4) + (lane & 15);
      __builtin_prefetch(A1 + (size_t)m * K1, 0, 0);
      if (K1 < K) __builtin_prefetch(A2 + (size_t)m * (K - K1), 0, 0);
    }
  }

  if (threadIdx.x < NOUT) blds[threadIdx.x] = bias[threadIdx.x];

  // Stage W -> LDS, swizzled into fragment order (vectorized source reads).
  for (int idx4 = threadIdx.x * 4; idx4 < K * NOUT; idx4 += 256 * 4) {
    v4f w4 = *(const v4f*)(W + idx4);
    const int k  = idx4 / NOUT;
    const int n0 = idx4 % NOUT;              // nn0 in {0,4,8,12}: no tile straddle
    const int kc = k >> 5, kk = k & 31;
    const int kh = (kk >> 3) & 1;
    const int fi = ((kk & 16) >> 1) | (kk & 7);
    const int ct = n0 >> 4, nn = n0 & 15;
    _Float16* tile = &wlds[(kc * NT + ct) * 512];
#pragma unroll
    for (int j = 0; j < 4; ++j)
      tile[(kh * 16 + nn + j) * 16 + fi] = (_Float16)w4[j];
  }
  __syncthreads();

  for (int mt = blockIdx.x * WAVES_PER_BLOCK + wave; mt < mtiles;
       mt += gridDim.x * WAVES_PER_BLOCK) {
    const int row0 = mt << 4;
    v16h af[NK];
#pragma unroll
    for (int kc = 0; kc < NK; ++kc)
      af[kc] = load_a_frag<K1, K>(A1, A2, row0, kc * 32);

    // Rolled: keeps exactly one accumulator set live; B fragments re-read
    // from LDS (dynamic ct => not hoistable/spillable).
#pragma clang loop unroll(disable)
    for (int ct = 0; ct < NT; ++ct) {
      const float bv = blds[ct * 16 + (lane & 15)];
      v8f acc;
#pragma unroll
      for (int r = 0; r < 8; ++r) acc[r] = bv;
#pragma unroll
      for (int kc = 0; kc < NK; ++kc) {
        v16h bf = *(const v16h*)(&wlds[(kc * NT + ct) * 512 + lane * 16]);
        acc = __builtin_amdgcn_wmma_f32_16x16x32_f16(
            /*neg_a=*/false, af[kc], /*neg_b=*/false, bf,
            /*c_mod=*/(short)0, acc, /*reuse_a=*/false, /*reuse_b=*/false);
      }
      const int col = ct * 16 + (lane & 15);
      const int rb  = row0 + ((lane >> 4) << 3);
#pragma unroll
      for (int r = 0; r < 8; ++r) {
        float v = acc[r];
        if (ACT == 1) v = v > 0.0f ? v : 0.0f;
        C[(size_t)(rb + r) * NOUT + col] = (TOUT)v;
      }
    }
  }
}

// ---------------------------------------------------------------------------
// Utility fill
// ---------------------------------------------------------------------------
__global__ void fill_kernel(float* __restrict__ p, float v, size_t n) {
  size_t t = (size_t)blockIdx.x * blockDim.x + threadIdx.x;
  size_t stride = (size_t)gridDim.x * blockDim.x;
  for (; t < n; t += stride) p[t] = v;
}

// ---------------------------------------------------------------------------
// Edge phase: alpha = <Q[dst], K[src]+rel_k[et]> / sqrt(D); segment max
// ---------------------------------------------------------------------------
__global__ void __launch_bounds__(256)
edge_alpha_kernel(const _Float16* __restrict__ Q, const _Float16* __restrict__ Kf,
                  const int* __restrict__ src, const int* __restrict__ dst,
                  const int* __restrict__ et, const float* __restrict__ relk,
                  float* __restrict__ alpha, float* __restrict__ amax, int E)
{
  int t = blockIdx.x * 256 + threadIdx.x;
  if (t >= E * 4) return;
  const int e = t >> 2, h = t & 3;
  const int s = src[e], d = dst[e], r = et[e];
  const _Float16* q = Q  + (size_t)d * 256 + h * 64;
  const _Float16* k = Kf + (size_t)s * 256 + h * 64;
  const float*   rk = relk + (size_t)r * 256 + h * 64;
  float acc = 0.0f;
#pragma unroll
  for (int c = 0; c < 8; ++c) {
    v8h qv = *(const v8h*)(q + c * 8);
    v8h kv = *(const v8h*)(k + c * 8);
    v4f r0 = *(const v4f*)(rk + c * 8);
    v4f r1 = *(const v4f*)(rk + c * 8 + 4);
#pragma unroll
    for (int i = 0; i < 4; ++i) {
      acc += (float)qv[i]     * ((float)kv[i]     + r0[i]);
      acc += (float)qv[4 + i] * ((float)kv[4 + i] + r1[i]);
    }
  }
  acc *= 0.125f;  // 1/sqrt(64)
  alpha[t] = acc;
  atomicMaxF(&amax[(size_t)d * 4 + h], acc);
}

// ea = exp(alpha - amax[dst]); denom[dst] += ea
__global__ void __launch_bounds__(256)
edge_exp_kernel(const int* __restrict__ dst, float* __restrict__ alpha,
                const float* __restrict__ amax, float* __restrict__ denom, int E)
{
  int t = blockIdx.x * 256 + threadIdx.x;
  if (t >= E * 4) return;
  const int e = t >> 2, h = t & 3;
  const int d = dst[e];
  float ea = __expf(alpha[t] - amax[(size_t)d * 4 + h]);
  alpha[t] = ea;
  atomicAdd(&denom[(size_t)d * 4 + h], ea);
}

// agg[dst] += (V[src] + rel_v[et]) * w  via packed 2xf16 atomics (CDNA5)
__global__ void __launch_bounds__(256)
edge_agg_kernel(const _Float16* __restrict__ V,
                const int* __restrict__ src, const int* __restrict__ dst,
                const int* __restrict__ et, const float* __restrict__ relv,
                const float* __restrict__ alpha, const float* __restrict__ denom,
                _Float16* __restrict__ agg, int E)
{
  int t = blockIdx.x * 256 + threadIdx.x;
  if (t >= E * 4) return;
  const int e = t >> 2, h = t & 3;
  const int s = src[e], d = dst[e], r = et[e];
  const float w = alpha[t] / (denom[(size_t)d * 4 + h] + 1e-16f);
  const _Float16* v = V + (size_t)s * 256 + h * 64;
  const float*   rv = relv + (size_t)r * 256 + h * 64;
  _Float16* o = agg + (size_t)d * 256 + h * 64;
#pragma unroll
  for (int c = 0; c < 8; ++c) {
    v8h vv = *(const v8h*)(v + c * 8);
    v4f r0 = *(const v4f*)(rv + c * 8);
    v4f r1 = *(const v4f*)(rv + c * 8 + 4);
    float f[8];
#pragma unroll
    for (int i = 0; i < 4; ++i) {
      f[i]     = ((float)vv[i]     + r0[i]) * w;
      f[4 + i] = ((float)vv[4 + i] + r1[i]) * w;
    }
#pragma unroll
    for (int j = 0; j < 4; ++j)
      atomic_pk_add_f16(o + c * 8 + j * 2, f[2 * j], f[2 * j + 1]);
  }
}

// ---------------------------------------------------------------------------
// y = LayerNorm(g*a + (1-g)*b), g = sigmoid(gate_lin). One wave per row
// (HID=64 -> 2 channels per lane), warp reductions via __shfl_xor (wave32).
// ---------------------------------------------------------------------------
__global__ void __launch_bounds__(256)
gate_ln_kernel(const float* __restrict__ a, const float* __restrict__ b,
               const float* __restrict__ gl, const float* __restrict__ gamma,
               const float* __restrict__ beta, float* __restrict__ y, int n)
{
  int row  = (blockIdx.x * 256 + threadIdx.x) >> 5;
  int lane = threadIdx.x & 31;
  if (row >= n) return;
  const size_t base = (size_t)row * 64;
  float v[2];
#pragma unroll
  for (int i = 0; i < 2; ++i) {
    int c = lane + i * 32;
    float g = 1.0f / (1.0f + __expf(-gl[base + c]));
    v[i] = g * a[base + c] + (1.0f - g) * b[base + c];
  }
  float s = v[0] + v[1];
#pragma unroll
  for (int o = 16; o > 0; o >>= 1) s += __shfl_xor(s, o, 32);
  const float mean = s * (1.0f / 64.0f);
  float d0 = v[0] - mean, d1 = v[1] - mean;
  float q = d0 * d0 + d1 * d1;
#pragma unroll
  for (int o = 16; o > 0; o >>= 1) q += __shfl_xor(q, o, 32);
  const float inv = rsqrtf(q * (1.0f / 64.0f) + 1e-5f);
#pragma unroll
  for (int i = 0; i < 2; ++i) {
    int c = lane + i * 32;
    y[base + c] = gamma[c] * ((v[i] - mean) * inv) + beta[c];
  }
}

// ---------------------------------------------------------------------------
// Host-side orchestration
// ---------------------------------------------------------------------------
extern "C" void kernel_launch(void* const* d_in, const int* in_sizes, int n_in,
                              void* d_out, int out_size, void* d_ws, size_t ws_size,
                              hipStream_t stream)
{
  (void)n_in; (void)out_size; (void)ws_size;
  const int N = in_sizes[0] / 256;   // IN = 256
  const int E = in_sizes[2];

  const float* x   = (const float*)d_in[0];
  const int*   ei  = (const int*)d_in[1];
  const int*   src = ei;
  const int*   dst = ei + E;
  const int*   et  = (const int*)d_in[2];

  auto P = [&](int i) { return (const float*)d_in[i]; };
  // params flattened depth-first in dict insertion order:
  // 3:embed.w 4:embed.b ; per layer l (base 5+20l):
  //  +0 Q.w +1 Q.b +2 K.w +3 K.b +4 V.w +5 V.b +6 rel_k +7 rel_v
  //  +8 out.w +9 out.b +10 gate.w +11 gate.b +12 ffn1.w +13 ffn1.b
  //  +14 ffn2.w +15 ffn2.b +16 n1.g +17 n1.b +18 n2.g +19 n2.b
  // 45:fusion1.w 46:b 47:fusion2.w 48:b 49:cls1.w 50:b 51:cls2.w 52:b
  const float* emb_w = P(3); const float* emb_b = P(4);

  // ---- workspace arena -----------------------------------------------------
  uintptr_t cur = (uintptr_t)d_ws;
  auto alloc = [&](size_t bytes) -> void* {
    uintptr_t p = (cur + 255) & ~(uintptr_t)255;
    cur = p + bytes;
    return (void*)p;
  };
  float* h_emb    = (float*)alloc((size_t)N * 64 * 4);
  float* houts[2] = { (float*)alloc((size_t)N * 64 * 4),
                      (float*)alloc((size_t)N * 64 * 4) };
  _Float16* Qb = (_Float16*)alloc((size_t)N * 256 * 2);
  _Float16* Kb = (_Float16*)alloc((size_t)N * 256 * 2);
  _Float16* Vb = (_Float16*)alloc((size_t)N * 256 * 2);
  _Float16* agg   = (_Float16*)alloc((size_t)N * 256 * 2);
  float* out_lin  = (float*)alloc((size_t)N * 64 * 4);
  float* gate_lin = (float*)alloc((size_t)N * 64 * 4);
  float* out_ln   = (float*)alloc((size_t)N * 64 * 4);
  float* ffn_out  = (float*)alloc((size_t)N * 64 * 4);
  float* alpha    = (float*)alloc((size_t)E * 4 * 4);
  float* amax     = (float*)alloc((size_t)N * 4 * 4);
  float* denom    = (float*)alloc((size_t)N * 4 * 4);
  _Float16* ffn_mid = Qb;  // alias: Q dead after edge phase

  const int gblk = ((N >> 4) + WAVES_PER_BLOCK - 1) / WAVES_PER_BLOCK;
  const int eblk = (E * 4 + 255) / 256;
  const int rblk = (N + 7) / 8;  // gate_ln: 8 rows per 256-thread block

  // ---- embed: h = relu(x @ We + be) ---------------------------------------
  gemm_kernel<float, float, 1, 256, 256, 64><<<gblk, 256, 0, stream>>>(
      x, x, emb_w, emb_b, h_emb, N);

  // ---- RGT layers ----------------------------------------------------------
  for (int l = 0; l < 2; ++l) {
    const int b = 5 + l * 20;
    const float* hin = (l == 0) ? h_emb : houts[l - 1];
    float* hnext = houts[l];

    // Q/K/V projections (f32 -> f16 buffers), K = 64, NOUT = 256
    gemm_kernel<float, _Float16, 0, 64, 64, 256><<<gblk, 256, 0, stream>>>(
        hin, hin, P(b + 0), P(b + 1), Qb, N);
    gemm_kernel<float, _Float16, 0, 64, 64, 256><<<gblk, 256, 0, stream>>>(
        hin, hin, P(b + 2), P(b + 3), Kb, N);
    gemm_kernel<float, _Float16, 0, 64, 64, 256><<<gblk, 256, 0, stream>>>(
        hin, hin, P(b + 4), P(b + 5), Vb, N);

    // segment-softmax attention over edges
    fill_kernel<<<2048, 256, 0, stream>>>(amax, -1e30f, (size_t)N * 4);
    fill_kernel<<<2048, 256, 0, stream>>>(denom, 0.0f, (size_t)N * 4);
    fill_kernel<<<2048, 256, 0, stream>>>((float*)agg, 0.0f, (size_t)N * 128);
    edge_alpha_kernel<<<eblk, 256, 0, stream>>>(Qb, Kb, src, dst, et,
                                                P(b + 6), alpha, amax, E);
    edge_exp_kernel<<<eblk, 256, 0, stream>>>(dst, alpha, amax, denom, E);
    edge_agg_kernel<<<eblk, 256, 0, stream>>>(Vb, src, dst, et, P(b + 7),
                                              alpha, denom, agg, E);

    // out-projection: agg[N,256] (f16) @ Wout[256,64]
    gemm_kernel<_Float16, float, 0, 256, 256, 64><<<gblk, 256, 0, stream>>>(
        agg, agg, P(b + 8), P(b + 9), out_lin, N);

    // gate 1: sigmoid([out, x] @ Wg), then LN(n1)
    gemm_kernel<float, float, 0, 64, 128, 64><<<gblk, 256, 0, stream>>>(
        out_lin, hin, P(b + 10), P(b + 11), gate_lin, N);
    gate_ln_kernel<<<rblk, 256, 0, stream>>>(out_lin, hin, gate_lin,
                                             P(b + 16), P(b + 17), out_ln, N);

    // FFN: 64 -> 256 (relu, f16 mid) -> 64
    gemm_kernel<float, _Float16, 1, 64, 64, 256><<<gblk, 256, 0, stream>>>(
        out_ln, out_ln, P(b + 12), P(b + 13), ffn_mid, N);
    gemm_kernel<_Float16, float, 0, 256, 256, 64><<<gblk, 256, 0, stream>>>(
        ffn_mid, ffn_mid, P(b + 14), P(b + 15), ffn_out, N);

    // gate 2: sigmoid([ffn, out_ln] @ Wg), then LN(n2) -> next h
    gemm_kernel<float, float, 0, 64, 128, 64><<<gblk, 256, 0, stream>>>(
        ffn_out, out_ln, P(b + 10), P(b + 11), gate_lin, N);
    gate_ln_kernel<<<rblk, 256, 0, stream>>>(ffn_out, out_ln, gate_lin,
                                             P(b + 18), P(b + 19), hnext, N);
  }

  // ---- fusion + classifier -------------------------------------------------
  float* t1 = out_lin;   // reuse dead buffers
  float* t2 = gate_lin;
  float* t3 = out_ln;
  gemm_kernel<float, float, 1, 64, 128, 64><<<gblk, 256, 0, stream>>>(
      houts[0], houts[1], P(45), P(46), t1, N);              // fusion1 + relu
  gemm_kernel<float, float, 0, 64, 64, 64><<<gblk, 256, 0, stream>>>(
      t1, t1, P(47), P(48), t2, N);                          // fusion2
  gemm_kernel<float, float, 1, 64, 64, 64><<<gblk, 256, 0, stream>>>(
      t2, t2, P(49), P(50), t3, N);                          // cls1 + relu
  gemm_kernel<float, float, 0, 64, 64, 16><<<gblk, 256, 0, stream>>>(
      t3, t3, P(51), P(52), (float*)d_out, N);               // cls2
}
